// PyTorchNeuralCDE_9466107921053
// MI455X (gfx1250) — matrix operand order repaired and checked by reading, not compiled, not run on hardware
//
#include <hip/hip_runtime.h>

// ---------------------------------------------------------------------------
// Neural CDE forward for MI455X (gfx1250).
// B=256, L=1024, I=16, H=128, W=512, O=4.
// Batch-partitioned persistent kernel: 16 workgroups x 16 batch rows,
// 512 threads (16 wave32 waves) each. All step GEMMs via
// v_wmma_f32_16x16x32_f16; fp32 recurrent state in LDS; fp16 weights
// streamed from L2 every step (2.6MB total -> L2 resident).
// A laundered (asm-opaque) integer offset is added to the weight pointers
// each step: blocks LICM from hoisting the t-invariant weight stream out of
// the 1023-step scan (previously caused scratch spills), while preserving
// pointer provenance so the loads stay on the global_load path (not flat).
// ---------------------------------------------------------------------------

typedef __attribute__((ext_vector_type(16))) _Float16 v16h;
typedef __attribute__((ext_vector_type(8)))  _Float16 v8h;
typedef __attribute__((ext_vector_type(8)))  float    v8f;

constexpr int B_  = 256;
constexpr int L_  = 1024;
constexpr int I_  = 16;
constexpr int H_  = 128;
constexpr int W_  = 512;
constexpr int O_  = 4;
constexpr int BT  = 16;    // batch tile per workgroup (== WMMA M)
constexpr int TPB = 512;   // 16 waves (wave32): 4 waves / SIMD32

// workspace layout (bytes)
constexpr size_t WS_WF1 = 0;                          // 128*512*2   = 128 KB
constexpr size_t WS_WF2 = 128u * 1024u;               // 512*512*2   = 512 KB
constexpr size_t WS_WF3 = 640u * 1024u;               // 512*2048*2  = 2 MB

__device__ __forceinline__ float fast_tanh(float x) {
  x = fminf(10.0f, fmaxf(-10.0f, x));
  float t = __builtin_amdgcn_exp2f(x * 2.8853900817779268f);  // e^(2x)
  return (t - 1.0f) * __builtin_amdgcn_rcpf(t + 1.0f);
}

__device__ __forceinline__ float fast_sigmoid(float x) {
  float e = __builtin_amdgcn_exp2f(-x * 1.4426950408889634f); // e^(-x)
  return __builtin_amdgcn_rcpf(1.0f + e);
}

// A-fragment (16x32 f16, M x K) from row-major fp16 LDS buffer.
// ISA 7.12.2: lane = row (lane&15), half (lane>>4)=kh: halfs cover
// K in [8kh, 8kh+8) then [16+8kh, 16+8kh+8)  -> two 16B LDS chunks.
__device__ __forceinline__ v16h load_a_frag(const _Float16* hbuf, int ldk,
                                            int kt, int lane) {
  const int row = lane & 15;
  const int kh  = lane >> 4;
  const _Float16* p = hbuf + row * ldk + (kt << 5) + (kh << 3);
  v8h lo = *(const v8h*)(p);
  v8h hi = *(const v8h*)(p + 16);
  return __builtin_shufflevector(lo, hi, 0, 1, 2, 3, 4, 5, 6, 7,
                                         8, 9, 10, 11, 12, 13, 14, 15);
}

// ---------------------------------------------------------------------------
// Weight prep: fp32 (K x N) row-major  ->  fp16 WMMA B-fragment layout.
// Fragment (kt,nt) = B tile 32x16; stored frag-contiguous (1KB each),
// frag index = nt*KT + kt (K innermost for the streaming K-loop).
// Per ISA: lane = col (lane&15), half kh: lane's 16 halfs = K in
// [kt*32 + 8*kh, +8) then [kt*32 + 16 + 8*kh, +8) -> 32B per lane.
// ---------------------------------------------------------------------------
__global__ void prep_wfrag(const float* __restrict__ src,
                           _Float16* __restrict__ dst, int K, int N) {
  int gid  = blockIdx.x * blockDim.x + threadIdx.x;
  int KT   = K >> 5;
  int NT   = N >> 4;
  int total = KT * NT * 32;
  if (gid >= total) return;
  int lane = gid & 31;
  int frag = gid >> 5;
  int kt   = frag % KT;
  int nt   = frag / KT;
  int n    = (nt << 4) + (lane & 15);
  int kb   = (kt << 5) + ((lane >> 4) << 3);
  _Float16* o = dst + ((size_t)frag << 9) + (lane << 4);
#pragma unroll
  for (int p = 0; p < 16; ++p)
    o[p] = (_Float16)src[(size_t)(kb + p + ((p >> 3) << 3)) * N + n];
}

// ---------------------------------------------------------------------------
// Main persistent kernel: one workgroup integrates 16 batch rows for all
// 1023 steps.
// ---------------------------------------------------------------------------
__global__ __launch_bounds__(TPB, 1)
void ncde_main(const float* __restrict__ ts, const float* __restrict__ x,
               const _Float16* __restrict__ wf1f_in,
               const _Float16* __restrict__ wf2f_in,
               const _Float16* __restrict__ wf3f_in,
               const float* __restrict__ bf1, const float* __restrict__ bf2,
               const float* __restrict__ bf3,
               const float* __restrict__ Wi1, const float* __restrict__ bi1,
               const float* __restrict__ Wi2, const float* __restrict__ bi2,
               const float* __restrict__ Wo1, const float* __restrict__ bo1,
               const float* __restrict__ Wo2, const float* __restrict__ bo2,
               float* __restrict__ out) {
  __shared__ float    yf32[BT * H_];     // fp32 master state
  __shared__ _Float16 yh[BT * H_];       // fp16 copy (WMMA A input)
  __shared__ _Float16 h1s[BT * W_];
  __shared__ _Float16 h2s[BT * W_];
  __shared__ float    dxs[BT * I_];
  __shared__ float    dtss[BT];
  __shared__ float    bf1s[W_];
  __shared__ float    bf2s[W_];
  __shared__ float    bf3s[H_ * I_];
  __shared__ float    headT[BT * 32];

  const int tid  = threadIdx.x;
  const int lane = tid & 31;
  const int wave = tid >> 5;             // 0..15
  const int b0   = blockIdx.x * BT;
  const int col  = lane & 15;
  const int kh   = lane >> 4;

  // stage biases
  for (int i = tid; i < W_; i += TPB) { bf1s[i] = bf1[i]; bf2s[i] = bf2[i]; }
  for (int i = tid; i < H_ * I_; i += TPB) bf3s[i] = bf3[i];

  // ---- y0 = tanh(x0 @ Wi1 + bi1) @ Wi2 + bi2  (tiny, VALU) ----
  for (int e = tid; e < BT * W_; e += TPB) {
    int r = e / W_, c = e % W_;
    const float* xr = x + (size_t)(b0 + r) * L_ * I_;  // x[b,0,:]
    float acc = bi1[c];
#pragma unroll
    for (int j = 0; j < I_; ++j) acc += xr[j] * Wi1[j * W_ + c];
    h1s[r * W_ + c] = (_Float16)fast_tanh(acc);
  }
  __syncthreads();
  for (int e = tid; e < BT * H_; e += TPB) {
    int r = e / H_, k = e % H_;
    float acc = bi2[k];
    for (int c = 0; c < W_; ++c)
      acc += (float)h1s[r * W_ + c] * Wi2[c * H_ + k];
    yf32[r * H_ + k] = acc;
    yh[r * H_ + k]   = (_Float16)acc;
  }
  __syncthreads();

  // ---- sequential scan: 1023 Euler steps ----
  for (int t = 0; t < L_ - 1; ++t) {
    // Launder a zero offset through asm so the weight-load addresses are not
    // provably t-invariant (blocks LICM -> no cross-step spill), while the
    // base pointers keep kernel-arg provenance (loads stay global_load).
    size_t wofs = 0;
    asm volatile("" : "+s"(wofs));
    const _Float16* wf1f = wf1f_in + wofs;
    const _Float16* wf2f = wf2f_in + wofs;
    const _Float16* wf3f = wf3f_in + wofs;

    // stage dx (16x16) and dt (16)
    if (tid < BT * I_) {
      int r = tid >> 4, j = tid & 15;
      const float* xp = x + ((size_t)(b0 + r) * L_ + t) * I_ + j;
      dxs[tid] = xp[I_] - xp[0];
      __builtin_prefetch(xp + 2 * I_, 0, 3);           // next step's x
      if (tid < BT) {
        const float* tp = ts + (size_t)(b0 + tid) * L_ + t;
        dtss[tid] = tp[1] - tp[0];
      }
    }
    __syncthreads();

    // phase 1: h1 = tanh(y @ Wf1 + bf1)   M=16, K=128, N=512 (2 n-tiles/wave)
    {
      v8f zero = {};
      v8f acc[2] = {zero, zero};
#pragma unroll 2
      for (int kt = 0; kt < H_ / 32; ++kt) {           // 4
        v16h a = load_a_frag(yh, H_, kt, lane);
#pragma unroll
        for (int q = 0; q < 2; ++q) {
          int nt = wave * 2 + q;
          v16h bf = *(const v16h*)(wf1f +
                      (((size_t)nt * (H_ / 32) + kt) << 9) + (lane << 4));
          acc[q] = __builtin_amdgcn_wmma_f32_16x16x32_f16(
              false, a, false, bf, (short)0, acc[q], false, false);
        }
      }
#pragma unroll
      for (int q = 0; q < 2; ++q) {
        int n = (wave * 2 + q) * 16 + col;
        float bias = bf1s[n];
#pragma unroll
        for (int g = 0; g < 8; ++g) {
          int row = g + (kh << 3);
          h1s[row * W_ + n] = (_Float16)fast_tanh(acc[q][g] + bias);
        }
      }
    }
    __syncthreads();

    // phase 2: h2 = tanh(h1 @ Wf2 + bf2)  M=16, K=512, N=512
    {
      v8f zero = {};
      v8f acc[2] = {zero, zero};
#pragma unroll 2
      for (int kt = 0; kt < W_ / 32; ++kt) {           // 16
        v16h a = load_a_frag(h1s, W_, kt, lane);
#pragma unroll
        for (int q = 0; q < 2; ++q) {
          int nt = wave * 2 + q;
          v16h bf = *(const v16h*)(wf2f +
                      (((size_t)nt * (W_ / 32) + kt) << 9) + (lane << 4));
          acc[q] = __builtin_amdgcn_wmma_f32_16x16x32_f16(
              false, a, false, bf, (short)0, acc[q], false, false);
        }
      }
#pragma unroll
      for (int q = 0; q < 2; ++q) {
        int n = (wave * 2 + q) * 16 + col;
        float bias = bf2s[n];
#pragma unroll
        for (int g = 0; g < 8; ++g) {
          int row = g + (kh << 3);
          h2s[row * W_ + n] = (_Float16)fast_tanh(acc[q][g] + bias);
        }
      }
    }
    __syncthreads();

    // phase 3: f = h2 @ Wf3 + bf3 (reshape B,H,I); dy = f . dx; y += dy*dt.
    // n-tile i of Wf3 == columns (i, j=0..15): each C tile is (16 batch x 16 j)
    // for a single i -> dy[b,i] = rowwise dot with dx. Wave owns i in
    // [wave*8, wave*8+8).
    {
      float dxv[8];
#pragma unroll
      for (int g = 0; g < 8; ++g)
        dxv[g] = dxs[(g + (kh << 3)) * I_ + col];

#pragma unroll 1
      for (int sb = 0; sb < 2; ++sb) {                 // 4 i's per block
        v8f zero = {};
        v8f acc[4] = {zero, zero, zero, zero};
#pragma unroll 2
        for (int kt = 0; kt < W_ / 32; ++kt) {         // 16
          v16h a = load_a_frag(h2s, W_, kt, lane);
#pragma unroll
          for (int q = 0; q < 4; ++q) {
            int i = wave * 8 + sb * 4 + q;
            v16h bf = *(const v16h*)(wf3f +
                        (((size_t)i * (W_ / 32) + kt) << 9) + (lane << 4));
            acc[q] = __builtin_amdgcn_wmma_f32_16x16x32_f16(
                false, a, false, bf, (short)0, acc[q], false, false);
          }
        }
#pragma unroll
        for (int q = 0; q < 4; ++q) {
          int i = wave * 8 + sb * 4 + q;
          float bias = bf3s[i * I_ + col];
#pragma unroll
          for (int g = 0; g < 8; ++g) {
            float v = (acc[q][g] + bias) * dxv[g];
            // reduce over the 16 j-lanes of each half-wave
#pragma unroll
            for (int m = 1; m < 16; m <<= 1) v += __shfl_xor(v, m, 32);
            if (col == 0) {
              int row = g + (kh << 3);
              float yv = yf32[row * H_ + i] + v * dtss[row];
              yf32[row * H_ + i] = yv;
              yh[row * H_ + i]   = (_Float16)yv;
            }
          }
        }
      }
    }
    __syncthreads();
  }

  // ---- output head: sigmoid(tanh(yT @ Wo1 + bo1) @ Wo2 + bo2) ----
  for (int e = tid; e < BT * 32; e += TPB) {
    int r = e >> 5, c = e & 31;
    float acc = bo1[c];
    for (int k = 0; k < H_; ++k) acc += yf32[r * H_ + k] * Wo1[k * 32 + c];
    headT[e] = fast_tanh(acc);
  }
  __syncthreads();
  if (tid < BT * O_) {
    int r = tid / O_, c = tid % O_;
    float acc = bo2[c];
#pragma unroll
    for (int k = 0; k < 32; ++k) acc += headT[r * 32 + k] * Wo2[k * O_ + c];
    out[(size_t)(b0 + r) * O_ + c] = fast_sigmoid(acc);
  }
}

extern "C" void kernel_launch(void* const* d_in, const int* in_sizes, int n_in,
                              void* d_out, int out_size, void* d_ws,
                              size_t ws_size, hipStream_t stream) {
  (void)in_sizes; (void)n_in; (void)out_size; (void)ws_size;
  const float* ts  = (const float*)d_in[0];
  const float* x   = (const float*)d_in[1];
  const float* Wi1 = (const float*)d_in[2];
  const float* bi1 = (const float*)d_in[3];
  const float* Wi2 = (const float*)d_in[4];
  const float* bi2 = (const float*)d_in[5];
  const float* Wf1 = (const float*)d_in[6];
  const float* bf1 = (const float*)d_in[7];
  const float* Wf2 = (const float*)d_in[8];
  const float* bf2 = (const float*)d_in[9];
  const float* Wf3 = (const float*)d_in[10];
  const float* bf3 = (const float*)d_in[11];
  const float* Wo1 = (const float*)d_in[12];
  const float* bo1 = (const float*)d_in[13];
  const float* Wo2 = (const float*)d_in[14];
  const float* bo2 = (const float*)d_in[15];
  float* out = (float*)d_out;

  char* ws = (char*)d_ws;
  _Float16* wf1f = (_Float16*)(ws + WS_WF1);
  _Float16* wf2f = (_Float16*)(ws + WS_WF2);
  _Float16* wf3f = (_Float16*)(ws + WS_WF3);

  // weight fragment prep (re-run every call: ws is poisoned by harness)
  prep_wfrag<<<(4 * 32 * 32 + 255) / 256, 256, 0, stream>>>(Wf1, wf1f, 128, 512);
  prep_wfrag<<<(16 * 32 * 32 + 255) / 256, 256, 0, stream>>>(Wf2, wf2f, 512, 512);
  prep_wfrag<<<(16 * 128 * 32 + 255) / 256, 256, 0, stream>>>(Wf3, wf3f, 512, 2048);

  ncde_main<<<B_ / BT, TPB, 0, stream>>>(ts, x, wf1f, wf2f, wf3f,
                                         bf1, bf2, bf3,
                                         Wi1, bi1, Wi2, bi2,
                                         Wo1, bo1, Wo2, bo2, out);
}
